// BiLSTMCRFNER_69690139345422
// MI455X (gfx1250) — compile-verified, matrix operand order checked by reading, not run on hardware
//
#include <hip/hip_runtime.h>
#include <hip/hip_bf16.h>

#define BB 64
#define SS 128
#define WW 16
#define CDIM 50
#define NF 50
#define WDIM 300
#define COMB 450
#define K0PAD 480
#define HID 256
#define NLAB 5
#define NTOK (BB * SS)   // 8192
#define NGATE 1024       // 4*HID
#define N2 2048          // both directions fused

typedef __attribute__((ext_vector_type(16))) _Float16 v16h;
typedef __attribute__((ext_vector_type(8)))  _Float16 v8h;
typedef __attribute__((ext_vector_type(8)))  float    v8f;
typedef __attribute__((ext_vector_type(4)))  unsigned int u32x4;
typedef __attribute__((ext_vector_type(8)))  int          i32x8;
typedef __attribute__((ext_vector_type(4)))  int          i32x4;

__device__ __forceinline__ v8f wmma_f16(v16h a, v16h b, v8f c) {
  // D = A(16x32 f16) * B(32x16 f16) + C(16x16 f32)
  return __builtin_amdgcn_wmma_f32_16x16x32_f16(
      /*neg_a=*/false, a, /*neg_b=*/false, b,
      /*c_mod=*/(short)0, c, /*reuse_a=*/false, /*reuse_b=*/false);
}

// Load a 16x32 f16 fragment from a row-major [rows][ldk] array of halves.
// Per ISA layout: lane = m + 16*ksel; halves j<8 -> K = ksel*8+j,
// j>=8 -> K = 16 + ksel*8 + (j-8). Two contiguous 16-byte loads per lane.
__device__ __forceinline__ v16h load_frag(const _Float16* __restrict__ p, int ldk,
                                          int row0, int k0, int lane) {
  const int r  = lane & 15;
  const int ks = (lane >> 4) & 1;
  const _Float16* q = p + (size_t)(row0 + r) * ldk + k0 + ks * 8;
  v8h lo = *(const v8h*)(q);
  v8h hi = *(const v8h*)(q + 16);
  v16h o;
#pragma unroll
  for (int i = 0; i < 8; ++i) { o[i] = lo[i]; o[i + 8] = hi[i]; }
  return o;
}

__device__ __forceinline__ float sigmf(float x) {
  return 1.f / (1.f + __expf(-x));
}

// ---------------- weight preprocessing ----------------
// dst[n*Kpad + k] = (k < K) ? src[k*N + n] : 0   (fp32 -> fp16, transpose to N-major)
__global__ void transpose_f2h(const float* __restrict__ src, _Float16* __restrict__ dst,
                              int K, int Kpad, int N) {
  int idx = blockIdx.x * blockDim.x + threadIdx.x;
  if (idx >= N * Kpad) return;
  int n = idx / Kpad, k = idx % Kpad;
  dst[idx] = (k < K) ? (_Float16)src[(size_t)k * N + n] : (_Float16)0.f;
}

__global__ void concat2(const float* __restrict__ a, const float* __restrict__ b,
                        float* __restrict__ dst, int n) {
  int i = blockIdx.x * blockDim.x + threadIdx.x;
  if (i < n) dst[i] = a[i];
  else if (i < 2 * n) dst[i] = b[i - n];
}

// ---------------- char CNN + embedding concat -> fp16 A0 ----------------
__global__ void charcnn_embed(const int* __restrict__ word_ids, const int* __restrict__ char_ids,
                              const float* __restrict__ word_emb, const float* __restrict__ char_emb,
                              const float* __restrict__ w3, const float* __restrict__ b3,
                              const float* __restrict__ w4, const float* __restrict__ b4,
                              const float* __restrict__ w5, const float* __restrict__ b5,
                              _Float16* __restrict__ A0h) {
  const int tok = blockIdx.x;
  const int tid = threadIdx.x;
  __shared__ float ce[WW][CDIM];
  for (int i = tid; i < WW * CDIM; i += blockDim.x) {
    int p = i / CDIM, c = i % CDIM;
    int cid = char_ids[tok * WW + p];
    ce[p][c] = char_emb[cid * CDIM + c];
  }
  const int wid = word_ids[tok];
  _Float16* dst = A0h + (size_t)tok * K0PAD;
  for (int c = tid; c < WDIM; c += blockDim.x)
    dst[c] = (_Float16)word_emb[(size_t)wid * WDIM + c];
  for (int c = COMB + tid; c < K0PAD; c += blockDim.x)
    dst[c] = (_Float16)0.f;   // K padding
  __syncthreads();
  if (tid < 3 * NF) {
    const int kind = tid / NF;
    const int f = tid % NF;
    const float* w; const float* bb; int k, pad, olen;
    if (kind == 0)      { w = w3; bb = b3; k = 3; pad = 1; olen = 16; }
    else if (kind == 1) { w = w4; bb = b4; k = 4; pad = 2; olen = 17; }
    else                { w = w5; bb = b5; k = 5; pad = 2; olen = 16; }
    float mx = -1e30f;
    for (int p = 0; p < olen; ++p) {
      float s = bb[f];
      for (int tap = 0; tap < k; ++tap) {
        int q = p - pad + tap;
        if (q >= 0 && q < WW) {
          const float* wp = w + (size_t)tap * CDIM * NF + f;
#pragma unroll 5
          for (int c = 0; c < CDIM; ++c) s += ce[q][c] * wp[c * NF];
        }
      }
      mx = fmaxf(mx, s);
    }
    dst[WDIM + kind * NF + f] = (_Float16)fmaxf(mx, 0.f);  // relu(max) == max(relu)
  }
}

// ---------------- input GEMM: xg[M][N2] = Ah[M][Kpad] * Bt[n][Kpad]^T + bias ----------------
// block = 256 threads = 8 waves; wave computes 32x32 (2x2 WMMA tiles); block tile 64x128.
// A block tile (64 x Kpad halves, <=64KB) is DMA'd into LDS once per block by the
// Tensor Data Mover (TENSORcnt); B streams from L2 via pipelined global_load_b128.
__global__ void __launch_bounds__(256) gemm_xg(const _Float16* __restrict__ Ah,
                                               const _Float16* __restrict__ Bt,
                                               const float* __restrict__ bias,
                                               float* __restrict__ Cg,
                                               int Kpad, int N) {
  __shared__ _Float16 Atile[64 * 512];          // 64KB max (Kpad<=512)
  const int lane = threadIdx.x & 31;
  const int wave = threadIdx.x >> 5;
  const int wm = wave & 1, wn = wave >> 1;
  const int m0blk = blockIdx.x * 64;
  const int n0 = blockIdx.y * 128 + wn * 32;

  if (wave == 0) {
    // Build Tensor DMA Descriptor (D#), 2-D tile: 64 rows x Kpad halves.
    unsigned long long ga = (unsigned long long)(uintptr_t)(Ah + (size_t)m0blk * Kpad);
    unsigned int lds = (unsigned int)(uintptr_t)(&Atile[0]);
    u32x4 g0;
    g0[0] = 1u;                                            // count=1 (valid), user mode
    g0[1] = lds;                                           // lds_addr (bytes)
    g0[2] = (unsigned int)(ga & 0xffffffffu);              // global_addr[31:0]
    g0[3] = (unsigned int)((ga >> 32) & 0x01ffffffu)       // global_addr[56:32]
            | (2u << 30);                                  // type=2 ("image")
    i32x8 g1;
    g1[0] = (int)(1u << 16);                               // data_size=1 -> 2 bytes
    g1[1] = (int)(((unsigned)Kpad & 0xffffu) << 16);       // tensor_dim0[15:0]
    g1[2] = (int)((((unsigned)Kpad >> 16) & 0xffffu)       // tensor_dim0[31:16]
                  | (64u << 16));                          // tensor_dim1[15:0] = 64
    g1[3] = (int)(((unsigned)Kpad & 0xffffu) << 16);       // tile_dim0 = Kpad
    g1[4] = 64;                                            // tile_dim1 = 64, tile_dim2 = 0
    g1[5] = Kpad;                                          // tensor_dim0_stride[31:0]
    g1[6] = 0;                                             // stride0 hi / stride1 lo
    g1[7] = 0;
    i32x4 z4 = {0, 0, 0, 0};
    i32x8 z8 = {0, 0, 0, 0, 0, 0, 0, 0};
    __builtin_amdgcn_tensor_load_to_lds(g0, g1, z4, z4, z8, 0);
    __builtin_amdgcn_s_wait_tensorcnt(0);
  }
  __syncthreads();   // publish LDS tile to all waves

  v8f z = {};
  v8f acc[2][2] = {{z, z}, {z, z}};
  const int m0l = wm * 32;                       // row within LDS tile
  for (int k0 = 0; k0 < Kpad; k0 += 32) {
    v16h a0 = load_frag(Atile, Kpad, m0l,      k0, lane);  // ds_load_b128 x2
    v16h a1 = load_frag(Atile, Kpad, m0l + 16, k0, lane);
    v16h b0 = load_frag(Bt, Kpad, n0,      k0, lane);      // global_load_b128 x2
    v16h b1 = load_frag(Bt, Kpad, n0 + 16, k0, lane);
    acc[0][0] = wmma_f16(a0, b0, acc[0][0]);
    acc[0][1] = wmma_f16(a0, b1, acc[0][1]);
    acc[1][0] = wmma_f16(a1, b0, acc[1][0]);
    acc[1][1] = wmma_f16(a1, b1, acc[1][1]);
  }
  const int nlo = lane & 15;
  const int mhi = (lane >> 4) * 8;
#pragma unroll
  for (int mt = 0; mt < 2; ++mt)
#pragma unroll
    for (int nt = 0; nt < 2; ++nt)
#pragma unroll
      for (int r = 0; r < 8; ++r) {
        int row = m0blk + wm * 32 + mt * 16 + mhi + r;
        int col = n0 + nt * 16 + nlo;
        Cg[(size_t)row * N + col] = acc[mt][nt][r] + bias[col];
      }
}

// ---------------- persistent LSTM recurrence (one block per direction) ----------------
// h (fp16) lives in LDS and is reloaded as WMMA A fragments each step; c lives in VGPRs
// (each lane permanently owns its (batch, hidden) elements via the C/D fragment layout).
__global__ void __launch_bounds__(1024) lstm_rec(const float* __restrict__ xg,
                                                 const _Float16* __restrict__ WhhT_all,
                                                 _Float16* __restrict__ outh,
                                                 float* __restrict__ outf,
                                                 int write_f32) {
  const int dir  = blockIdx.x;
  const int lane = threadIdx.x & 31;
  const int wave = threadIdx.x >> 5;   // 0..31
  __shared__ _Float16 hH[BB * HID];    // 32 KB
  for (int i = threadIdx.x; i < BB * HID; i += blockDim.x) hH[i] = (_Float16)0.f;
  __syncthreads();
  const _Float16* Wd = WhhT_all + (size_t)dir * NGATE * HID;  // [n=1024][k=256] halves
  const int nlo = lane & 15;
  const int mhi = (lane >> 4) * 8;
  int m0[2], hu0[2];
#pragma unroll
  for (int ti = 0; ti < 2; ++ti) {
    int tid = wave * 2 + ti;           // 0..63 -> 4 m-tiles x 16 hu-tiles
    m0[ti]  = (tid >> 4) * 16;
    hu0[ti] = (tid & 15) * 16;
  }
  float creg[2][8];
#pragma unroll
  for (int ti = 0; ti < 2; ++ti)
#pragma unroll
    for (int r = 0; r < 8; ++r) creg[ti][r] = 0.f;

  for (int t = 0; t < SS; ++t) {
    const int s = dir ? (SS - 1 - t) : t;
    v8f acc[2][4];
#pragma unroll
    for (int ti = 0; ti < 2; ++ti)
#pragma unroll
      for (int g = 0; g < 4; ++g) {
        v8f a;
#pragma unroll
        for (int r = 0; r < 8; ++r) {
          int batch = m0[ti] + mhi + r;
          a[r] = xg[(size_t)(batch * SS + s) * N2 + dir * NGATE + g * HID + hu0[ti] + nlo];
        }
        acc[ti][g] = a;
      }
    for (int k0 = 0; k0 < HID; k0 += 32) {
#pragma unroll
      for (int ti = 0; ti < 2; ++ti) {
        v16h a = load_frag(hH, HID, m0[ti], k0, lane);
#pragma unroll
        for (int g = 0; g < 4; ++g) {
          v16h b = load_frag(Wd, HID, g * HID + hu0[ti], k0, lane);
          acc[ti][g] = wmma_f16(a, b, acc[ti][g]);
        }
      }
    }
    __syncthreads();  // all reads of h_{t-1} complete
#pragma unroll
    for (int ti = 0; ti < 2; ++ti)
#pragma unroll
      for (int r = 0; r < 8; ++r) {
        float iv = sigmf(acc[ti][0][r]);
        float fv = sigmf(acc[ti][1][r]);
        float gv = tanhf(acc[ti][2][r]);
        float ov = sigmf(acc[ti][3][r]);
        float c  = fv * creg[ti][r] + iv * gv;
        creg[ti][r] = c;
        float h = ov * tanhf(c);
        int batch = m0[ti] + mhi + r;
        int hu    = hu0[ti] + nlo;
        hH[batch * HID + hu] = (_Float16)h;
        size_t orow = (size_t)(batch * SS + s);
        if (write_f32) outf[orow * 512 + dir * HID + hu] = h;
        else           outh[orow * 512 + dir * HID + hu] = (_Float16)h;
      }
    __syncthreads();  // h_t visible before next step's fragment loads
  }
}

// ---------------- emissions (tiny GEMM, VALU) ----------------
__global__ void emissions_k(const float* __restrict__ x2, const float* __restrict__ ow,
                            const float* __restrict__ ob, float* __restrict__ emis) {
  int idx = blockIdx.x * blockDim.x + threadIdx.x;
  if (idx >= NTOK * NLAB) return;
  int tok = idx / NLAB, l = idx % NLAB;
  const float* xr = x2 + (size_t)tok * 512;
  float sacc = ob[l];
  for (int k = 0; k < 512; ++k) sacc += xr[k] * ow[k * NLAB + l];
  emis[idx] = sacc;
}

// ---------------- CRF loss (one thread per batch, deterministic reduce) ----------------
__global__ void crf_loss(const float* __restrict__ emis, const int* __restrict__ labels,
                         const int* __restrict__ lengths, const float* __restrict__ cs,
                         const float* __restrict__ cef, const float* __restrict__ ct,
                         float* __restrict__ out) {
  __shared__ float nll[BB];
  int b = threadIdx.x;
  if (b < BB) {
    const float* em = emis + (size_t)b * SS * NLAB;
    const int* tg = labels + b * SS;
    int L = lengths[b];
    // numerator
    float num = cs[tg[0]] + em[tg[0]];
    for (int t = 1; t < SS; ++t)
      if (t < L) num += ct[tg[t - 1] * NLAB + tg[t]] + em[t * NLAB + tg[t]];
    num += cef[tg[L - 1]];
    // denominator: forward algorithm
    float alpha[NLAB];
    for (int j = 0; j < NLAB; ++j) alpha[j] = cs[j] + em[j];
    for (int t = 1; t < SS; ++t) {
      if (t < L) {
        float na[NLAB];
        for (int j = 0; j < NLAB; ++j) {
          float m = -1e30f;
          for (int i = 0; i < NLAB; ++i) m = fmaxf(m, alpha[i] + ct[i * NLAB + j]);
          float ssum = 0.f;
          for (int i = 0; i < NLAB; ++i) ssum += __expf(alpha[i] + ct[i * NLAB + j] - m);
          na[j] = m + __logf(ssum) + em[t * NLAB + j];
        }
        for (int j = 0; j < NLAB; ++j) alpha[j] = na[j];
      }
    }
    float m = -1e30f;
    for (int j = 0; j < NLAB; ++j) m = fmaxf(m, alpha[j] + cef[j]);
    float ssum = 0.f;
    for (int j = 0; j < NLAB; ++j) ssum += __expf(alpha[j] + cef[j] - m);
    float denom = m + __logf(ssum);
    nll[b] = denom - num;  // -llh
  }
  __syncthreads();
  if (threadIdx.x == 0) {
    float sacc = 0.f;
    for (int i = 0; i < BB; ++i) sacc += nll[i];
    out[0] = sacc / (float)BB;
  }
}

extern "C" void kernel_launch(void* const* d_in, const int* in_sizes, int n_in,
                              void* d_out, int out_size, void* d_ws, size_t ws_size,
                              hipStream_t stream) {
  (void)in_sizes; (void)n_in; (void)out_size; (void)ws_size;
  const int*   word_ids  = (const int*)d_in[0];
  const int*   char_ids  = (const int*)d_in[1];
  const int*   labels    = (const int*)d_in[2];
  const int*   lengths   = (const int*)d_in[3];
  const float* word_emb  = (const float*)d_in[4];
  const float* char_emb  = (const float*)d_in[5];
  const float* conv_w3   = (const float*)d_in[6];
  const float* conv_b3   = (const float*)d_in[7];
  const float* conv_w4   = (const float*)d_in[8];
  const float* conv_b4   = (const float*)d_in[9];
  const float* conv_w5   = (const float*)d_in[10];
  const float* conv_b5   = (const float*)d_in[11];
  const float* out_w     = (const float*)d_in[12];
  const float* out_b     = (const float*)d_in[13];
  const float* crf_start = (const float*)d_in[14];
  const float* crf_end   = (const float*)d_in[15];
  const float* crf_trans = (const float*)d_in[16];
  // lstm order: l0f(Wih,Whh,b), l0r, l1f, l1r
  const float* Wih[4] = {(const float*)d_in[17], (const float*)d_in[20],
                         (const float*)d_in[23], (const float*)d_in[26]};
  const float* Whh[4] = {(const float*)d_in[18], (const float*)d_in[21],
                         (const float*)d_in[24], (const float*)d_in[27]};
  const float* bih[4] = {(const float*)d_in[19], (const float*)d_in[22],
                         (const float*)d_in[25], (const float*)d_in[28]};

  char* p = (char*)d_ws;
  auto take = [&](size_t bytes) -> char* {
    char* r = p; p += (bytes + 255) & ~(size_t)255; return r;
  };
  _Float16* A0h   = (_Float16*)take((size_t)NTOK * K0PAD * 2);
  _Float16* B0t   = (_Float16*)take((size_t)N2 * K0PAD * 2);
  float*    bias0 = (float*)   take((size_t)N2 * 4);
  _Float16* B1t   = (_Float16*)take((size_t)N2 * 512 * 2);
  float*    bias1 = (float*)   take((size_t)N2 * 4);
  _Float16* WhhT  = (_Float16*)take((size_t)4 * NGATE * HID * 2);
  _Float16* A1h   = (_Float16*)take((size_t)NTOK * 512 * 2);
  float*    xg    = (float*)   take((size_t)NTOK * N2 * 4);
  float*    x2    = (float*)   take((size_t)NTOK * 512 * 4);
  float*    emis  = (float*)   take((size_t)NTOK * NLAB * 4);

  const int TPB = 256;
  // weight preprocessing: transpose + fp16 convert
  {
    int n0c = NGATE * K0PAD;
    transpose_f2h<<<(n0c + TPB - 1) / TPB, TPB, 0, stream>>>(Wih[0], B0t, COMB, K0PAD, NGATE);
    transpose_f2h<<<(n0c + TPB - 1) / TPB, TPB, 0, stream>>>(Wih[1], B0t + (size_t)NGATE * K0PAD,
                                                             COMB, K0PAD, NGATE);
    int n1c = NGATE * 512;
    transpose_f2h<<<(n1c + TPB - 1) / TPB, TPB, 0, stream>>>(Wih[2], B1t, 512, 512, NGATE);
    transpose_f2h<<<(n1c + TPB - 1) / TPB, TPB, 0, stream>>>(Wih[3], B1t + (size_t)NGATE * 512,
                                                             512, 512, NGATE);
    int nhc = NGATE * HID;
    for (int dl = 0; dl < 4; ++dl)
      transpose_f2h<<<(nhc + TPB - 1) / TPB, TPB, 0, stream>>>(
          Whh[dl], WhhT + (size_t)dl * NGATE * HID, HID, HID, NGATE);
    concat2<<<(2 * NGATE + TPB - 1) / TPB, TPB, 0, stream>>>(bih[0], bih[1], bias0, NGATE);
    concat2<<<(2 * NGATE + TPB - 1) / TPB, TPB, 0, stream>>>(bih[2], bih[3], bias1, NGATE);
  }
  // features -> fp16 A0
  charcnn_embed<<<NTOK, 256, 0, stream>>>(word_ids, char_ids, word_emb, char_emb,
                                          conv_w3, conv_b3, conv_w4, conv_b4,
                                          conv_w5, conv_b5, A0h);
  // layer 0
  gemm_xg<<<dim3(NTOK / 64, N2 / 128), 256, 0, stream>>>(A0h, B0t, bias0, xg, K0PAD, N2);
  lstm_rec<<<2, 1024, 0, stream>>>(xg, WhhT, A1h, x2, 0);
  // layer 1 (xg buffer reused)
  gemm_xg<<<dim3(NTOK / 64, N2 / 128), 256, 0, stream>>>(A1h, B1t, bias1, xg, 512, N2);
  lstm_rec<<<2, 1024, 0, stream>>>(xg, WhhT + (size_t)2 * NGATE * HID, A1h, x2, 1);
  // emissions + CRF
  emissions_k<<<(NTOK * NLAB + TPB - 1) / TPB, TPB, 0, stream>>>(x2, out_w, out_b, emis);
  crf_loss<<<1, BB, 0, stream>>>(emis, labels, lengths, crf_start, crf_end, crf_trans,
                                 (float*)d_out);
}
